// ProposalLayer_89928025243735
// MI455X (gfx1250) — matrix operand order repaired and checked by reading, not compiled, not run on hardware
//
#include <hip/hip_runtime.h>
#include <hip/hip_bf16.h>

// ---------------------------------------------------------------------------
// Problem constants (match reference)
// ---------------------------------------------------------------------------
#define Bn   8
#define Hn   512
#define Wn   512
#define Cn   32
#define NPn  256
#define CROPn 16
#define CCn  64
#define NCLS 3

#define Mrows   (Bn * NPn)        // 2048 crops
#define R1rows  (Mrows * 64)      // conv1 GEMM rows (8x8 positions per crop)
#define K1dim   288               // 3*3*32
#define N1dim   64
#define R2rows  (Mrows * 16)      // conv2 GEMM rows (4x4 positions per crop)
#define K2dim   576               // 3*3*64
#define N2dim   128
#define KF1     2048              // 4*4*128 flatten
#define NFC     512

typedef _Float16 half8 __attribute__((ext_vector_type(8)));
typedef __attribute__((ext_vector_type(16))) _Float16 v16h;
typedef __attribute__((ext_vector_type(8)))  float    v8f;
typedef int i32x4 __attribute__((ext_vector_type(4)));

// Async global->LDS staging (CDNA5 ASYNCcnt path), guarded so missing builtins
// fall back to global_load_b128 + ds_store_b128.
#if defined(__has_builtin)
# if __has_builtin(__builtin_amdgcn_global_load_async_to_lds_b128)
#  define HAVE_ASYNC_LDS 1
# endif
#endif
#ifndef HAVE_ASYNC_LDS
# define HAVE_ASYNC_LDS 0
#endif
#define AS_GLOBAL __attribute__((address_space(1)))
#define AS_LDS    __attribute__((address_space(3)))

// ---------------------------------------------------------------------------
// 1) init per-segment stats
// ---------------------------------------------------------------------------
__global__ void ws_init_kernel(int* mny, int* mnx, int* mxy, int* mxx,
                               unsigned long long* swy, unsigned long long* swx,
                               unsigned long long* sw) {
  int i = blockIdx.x * 256 + threadIdx.x;
  if (i >= Bn * NPn) return;
  mny[i] = 0x7FFFFFFF; mnx[i] = 0x7FFFFFFF;
  mxy[i] = (int)0x80000000; mxx[i] = (int)0x80000000;
  swy[i] = 0ull; swx[i] = 0ull; sw[i] = 0ull;
}

// ---------------------------------------------------------------------------
// 2) segment reductions (deterministic: int atomics + fixed-point u64 sums)
// ---------------------------------------------------------------------------
__global__ void seg_reduce_kernel(const int* __restrict__ lab,
                                  const float* __restrict__ sz,
                                  const float* __restrict__ wt,
                                  int* mny, int* mnx, int* mxy, int* mxx,
                                  unsigned long long* swy,
                                  unsigned long long* swx,
                                  unsigned long long* sw) {
  int idx = blockIdx.x * 256 + threadIdx.x;
  if (idx >= Bn * Hn * Wn) return;
  int b   = idx >> 18;          // H*W = 2^18
  int pix = idx & 262143;
  int row = pix >> 9;           // W = 512
  int col = pix & 511;
  int s = b * NPn + lab[idx];
  atomicMin(&mny[s], row);
  atomicMin(&mnx[s], col);
  atomicMax(&mxy[s], row);
  atomicMax(&mxx[s], col);
  float w  = wt[idx];
  float sy = sz[(size_t)idx * 2 + 0];
  float sx = sz[(size_t)idx * 2 + 1];
  atomicAdd(&swy[s], (unsigned long long)(sy * w * 1048576.0f + 0.5f));
  atomicAdd(&swx[s], (unsigned long long)(sx * w * 1048576.0f + 0.5f));
  atomicAdd(&sw [s], (unsigned long long)(w      * 1048576.0f + 0.5f));
}

// ---------------------------------------------------------------------------
// 3) boxes: segment bbox -> recentered box with guessed sizes
// ---------------------------------------------------------------------------
__global__ void boxes_kernel(const int* mny, const int* mnx,
                             const int* mxy, const int* mxx,
                             const unsigned long long* swy,
                             const unsigned long long* swx,
                             const unsigned long long* sw,
                             float* __restrict__ nbb, float* __restrict__ out) {
  int i = blockIdx.x * 256 + threadIdx.x;
  if (i >= Bn * NPn) return;
  float y0 = (float)mny[i] * (1.0f / Hn);
  float x0 = (float)mnx[i] * (1.0f / Wn);
  float y1 = (float)(mxy[i] + 1) * (1.0f / Hn);
  float x1 = (float)(mxx[i] + 1) * (1.0f / Wn);
  float cy = 0.5f * (y0 + y1);
  float cx = 0.5f * (x0 + x1);
  const float inv = 1.0f / 1048576.0f;
  float den = (float)sw[i] * inv + 1e-8f;
  float sh = ((float)swy[i] * inv) / den * (1.0f / Hn);
  float swd = ((float)swx[i] * inv) / den * (1.0f / Wn);
  float b0 = cy - sh * 0.5f, b1 = cx - swd * 0.5f;
  float b2 = cy + sh * 0.5f, b3 = cx + swd * 0.5f;
  nbb[i * 4 + 0] = b0; nbb[i * 4 + 1] = b1;
  nbb[i * 4 + 2] = b2; nbb[i * 4 + 3] = b3;
  out[i * 4 + 0] = b0; out[i * 4 + 1] = b1;
  out[i * 4 + 2] = b2; out[i * 4 + 3] = b3;
}

// ---------------------------------------------------------------------------
// 4) fp32 -> f16 weight conversion
// ---------------------------------------------------------------------------
__global__ void cvt_f32_f16_kernel(const float* __restrict__ s,
                                   _Float16* __restrict__ d, int n) {
  int i = blockIdx.x * 256 + threadIdx.x;
  if (i < n) d[i] = (_Float16)s[i];
}

// ---------------------------------------------------------------------------
// 5) bilinear crops: [2048,16,16,32] f16
// ---------------------------------------------------------------------------
__global__ __launch_bounds__(256) void crop_kernel(const float* __restrict__ feat,
                                                   const float* __restrict__ nbb,
                                                   _Float16* __restrict__ crops) {
  int cr = blockIdx.x;                  // 0..2047
  int tid = threadIdx.x;
  int y = tid >> 4, x = tid & 15;
  float b0 = nbb[cr * 4 + 0], b1 = nbb[cr * 4 + 1];
  float b2 = nbb[cr * 4 + 2], b3 = nbb[cr * 4 + 3];
  float ly = (float)y * (1.0f / (CROPn - 1));
  float lx = (float)x * (1.0f / (CROPn - 1));
  float py = (b0 + (b2 - b0) * ly) * (Hn - 1);
  float px = (b1 + (b3 - b1) * lx) * (Wn - 1);
  py = fminf(fmaxf(py, 0.0f), (float)(Hn - 1));
  px = fminf(fmaxf(px, 0.0f), (float)(Wn - 1));
  int y0i = (int)floorf(py); y0i = y0i < 0 ? 0 : (y0i > Hn - 2 ? Hn - 2 : y0i);
  int x0i = (int)floorf(px); x0i = x0i < 0 ? 0 : (x0i > Wn - 2 ? Wn - 2 : x0i);
  float fy = py - (float)y0i, fx = px - (float)x0i;
  int bidx = cr >> 8;                   // NP = 256
  const float* f00 = feat + (((size_t)bidx * Hn + y0i) * Wn + x0i) * Cn;
  const float* f01 = f00 + Cn;
  const float* f10 = f00 + (size_t)Wn * Cn;
  const float* f11 = f10 + Cn;
  float w00 = (1.0f - fy) * (1.0f - fx), w01 = (1.0f - fy) * fx;
  float w10 = fy * (1.0f - fx),          w11 = fy * fx;
  _Float16* dst = crops + (((size_t)cr * CROPn + y) * CROPn + x) * Cn;
#pragma unroll
  for (int c = 0; c < Cn; ++c)
    dst[c] = (_Float16)(f00[c] * w00 + f01[c] * w01 + f10[c] * w10 + f11[c] * w11);
}

// ---------------------------------------------------------------------------
// 6) im2col for conv1 (16x16x32 -> rows 64 x K 288, SAME pad_top=0, stride 2)
// ---------------------------------------------------------------------------
__global__ void im2col1_kernel(const _Float16* __restrict__ crops,
                               _Float16* __restrict__ col) {
  int idx = blockIdx.x * 256 + threadIdx.x;     // R1rows * 9
  if (idx >= R1rows * 9) return;
  int r = idx / 9, t = idx - r * 9;
  int crop = r >> 6, pos = r & 63;
  int oy = pos >> 3, ox = pos & 7;
  int ky = t / 3, kx = t - ky * 3;
  int iy = oy * 2 + ky, ix = ox * 2 + kx;
  _Float16* dst = col + (size_t)r * K1dim + t * 32;
  if (iy < CROPn && ix < CROPn) {
    const _Float16* src = crops + (((size_t)crop * CROPn + iy) * CROPn + ix) * Cn;
#pragma unroll
    for (int j = 0; j < 4; ++j)
      *(half8*)(dst + j * 8) = *(const half8*)(src + j * 8);
  } else {
    half8 z = {};
#pragma unroll
    for (int j = 0; j < 4; ++j) *(half8*)(dst + j * 8) = z;
  }
}

// ---------------------------------------------------------------------------
// 7) im2col for conv2 (8x8x64 -> rows 16 x K 576)
// ---------------------------------------------------------------------------
__global__ void im2col2_kernel(const _Float16* __restrict__ c1out,
                               _Float16* __restrict__ col) {
  int idx = blockIdx.x * 256 + threadIdx.x;     // R2rows * 9
  if (idx >= R2rows * 9) return;
  int r = idx / 9, t = idx - r * 9;
  int crop = r >> 4, pos = r & 15;
  int oy = pos >> 2, ox = pos & 3;
  int ky = t / 3, kx = t - ky * 3;
  int iy = oy * 2 + ky, ix = ox * 2 + kx;
  _Float16* dst = col + (size_t)r * K2dim + t * 64;
  if (iy < 8 && ix < 8) {
    const _Float16* src = c1out + (((size_t)crop * 8 + iy) * 8 + ix) * CCn;
#pragma unroll
    for (int j = 0; j < 8; ++j)
      *(half8*)(dst + j * 8) = *(const half8*)(src + j * 8);
  } else {
    half8 z = {};
#pragma unroll
    for (int j = 0; j < 8; ++j) *(half8*)(dst + j * 8) = z;
  }
}

// ---------------------------------------------------------------------------
// 8) Tiled WMMA GEMM:  C[M,N] = relu(A[M,K] @ B[K,N] + bias)
//    f16 in / f32 accumulate / f16 out. Block tile 128xNT (NT=64 or 128),
//    8 wave32 waves; each wave 32 x NT/2 via 2 x (NT/32) v_wmma_f32_16x16x32_f16.
//    Double-buffered LDS: stage K-step kb+1 while computing kb (1 barrier/iter).
//    A tile staged via global_load_async_to_lds_b128 (ASYNCcnt) when available.
//    Requires M%128==0, N%NT==0, K%32==0 (true for all call sites).
// ---------------------------------------------------------------------------
template <int NT>
__global__ __launch_bounds__(256) void gemm_f16_wmma(
    const _Float16* __restrict__ A, const _Float16* __restrict__ Bm,
    const float* __restrict__ bias, _Float16* __restrict__ Cm,
    int M, int N, int K, int relu) {
  constexpr int NI = NT / 32;            // B 16x16 sub-tiles per wave (2 or 4)
  __shared__ _Float16 ldsA[2][128 * 40]; // rows padded 32->40 halves (bank spread)
  __shared__ _Float16 ldsB[2][NT * 40];  // stored transposed: ldsB[n][k]

  const int tid  = threadIdx.x;
  const int lane = tid & 31;
  const int wave = tid >> 5;
  const int waveM = wave >> 1;           // 0..3 -> 32-row strip
  const int waveN = wave & 1;            // 0..1 -> NT/2-col strip
  const int rowBase = blockIdx.x * 128;
  const int colBase = blockIdx.y * NT;

  v8f acc[2][NI] = {};

  const int arow = tid >> 1;             // 0..127
  const int akc  = (tid & 1) * 16;       // 0 / 16
  const int bk   = tid >> 3;             // 0..31
  const int bn0  = (tid & 7) * (NT / 8); // 0..NT-NT/8
  const int hi   = lane >> 4;
  const int l15  = lane & 15;
  const int ksteps = K >> 5;

  auto stageA = [&](int kb, int b) {
    const size_t g = (size_t)(rowBase + arow) * K + (size_t)kb * 32 + akc;
#if HAVE_ASYNC_LDS
    __builtin_amdgcn_global_load_async_to_lds_b128(
        (AS_GLOBAL i32x4*)(A + g),
        (AS_LDS i32x4*)(&ldsA[b][arow * 40 + akc]), 0, 0);
    __builtin_amdgcn_global_load_async_to_lds_b128(
        (AS_GLOBAL i32x4*)(A + g + 8),
        (AS_LDS i32x4*)(&ldsA[b][arow * 40 + akc + 8]), 0, 0);
#else
    half8 a0 = *(const half8*)(A + g);
    half8 a1 = *(const half8*)(A + g + 8);
    *(half8*)(&ldsA[b][arow * 40 + akc])     = a0;
    *(half8*)(&ldsA[b][arow * 40 + akc + 8]) = a1;
    if (kb + 2 < ksteps) __builtin_prefetch(A + g + 64, 0, 1);
#endif
  };
  auto stageB = [&](int kb, int b) {
    const size_t g = (size_t)(kb * 32 + bk) * N + colBase + bn0;
#pragma unroll
    for (int c = 0; c < NT / 64; ++c) {
      half8 bv = *(const half8*)(Bm + g + c * 8);
#pragma unroll
      for (int j = 0; j < 8; ++j)
        ldsB[b][(bn0 + c * 8 + j) * 40 + bk] = bv[j];
    }
  };
  auto compute = [&](int b) {
    // Fragment layouts per CDNA5 ISA 7.12.2:
    // A: halves 0-7 <- K = hi*8..+7 ; halves 8-15 <- K = 16+hi*8..+7
    // B (transposed in LDS): halves 0-7 <- K = hi*8..+7 ; halves 8-15 <- K = 16+hi*8..+7 at column n
    v16h afrag[2];
#pragma unroll
    for (int mi = 0; mi < 2; ++mi) {
      const int row = waveM * 32 + mi * 16 + l15;
      half8 lo = *(const half8*)(&ldsA[b][row * 40 + hi * 8]);
      half8 hv = *(const half8*)(&ldsA[b][row * 40 + 16 + hi * 8]);
      afrag[mi] = __builtin_shufflevector(lo, hv,
          0,1,2,3,4,5,6,7,8,9,10,11,12,13,14,15);
    }
#pragma unroll
    for (int ni = 0; ni < NI; ++ni) {
      const int n = waveN * (NT / 2) + ni * 16 + l15;
      half8 lo = *(const half8*)(&ldsB[b][n * 40 + hi * 8]);
      half8 hv = *(const half8*)(&ldsB[b][n * 40 + 16 + hi * 8]);
      v16h bfrag = __builtin_shufflevector(lo, hv,
          0,1,2,3,4,5,6,7,8,9,10,11,12,13,14,15);
#pragma unroll
      for (int mi = 0; mi < 2; ++mi)
        acc[mi][ni] = __builtin_amdgcn_wmma_f32_16x16x32_f16(
            false, afrag[mi], false, bfrag, (short)0, acc[mi][ni], false, false);
    }
  };

  stageA(0, 0);
  stageB(0, 0);
  int buf = 0;
  for (int kb = 0; kb < ksteps; ++kb) {
#if HAVE_ASYNC_LDS
    asm volatile("s_wait_asynccnt 0x0" ::: "memory");
#endif
    __syncthreads();
    if (kb + 1 < ksteps) {
      stageA(kb + 1, buf ^ 1);
      stageB(kb + 1, buf ^ 1);
    }
    compute(buf);
    buf ^= 1;
  }

  // epilogue: C layout VGPR r -> row r + hi*8, col = lane&15
#pragma unroll
  for (int mi = 0; mi < 2; ++mi)
#pragma unroll
    for (int ni = 0; ni < NI; ++ni) {
      const int gcol = colBase + waveN * (NT / 2) + ni * 16 + l15;
      const float bv = bias[gcol];
      const unsigned rb = (unsigned)(rowBase + waveM * 32 + mi * 16 + hi * 8);
#pragma unroll
      for (int r = 0; r < 8; ++r) {
        float v = acc[mi][ni][r] + bv;
        if (relu) v = fmaxf(v, 0.0f);
        Cm[(size_t)(rb + r) * (unsigned)N + (unsigned)gcol] = (_Float16)v;
      }
    }
}

// ---------------------------------------------------------------------------
// 9) heads: class-selected reg/score + recover boxes + final outputs
// ---------------------------------------------------------------------------
__global__ __launch_bounds__(64) void heads_kernel(
    const _Float16* __restrict__ x2, const float* __restrict__ reg_w,
    const float* __restrict__ reg_b, const float* __restrict__ score_w,
    const float* __restrict__ score_b, const int* __restrict__ cls,
    const float* __restrict__ nbb, float* __restrict__ out) {
  __shared__ float red[336];
  const int m = blockIdx.x;             // 0..2047
  const int tid = threadIdx.x;
  const int c = cls[m >> 8];
  const int c0 = c * 4;
  const _Float16* xr = x2 + (size_t)m * NFC;
  float a0 = 0, a1 = 0, a2 = 0, a3 = 0, a4 = 0;
  for (int k = tid; k < NFC; k += 64) {
    float xv = (float)xr[k];
    const float* rw = reg_w + (size_t)k * (4 * NCLS) + c0;
    a0 += xv * rw[0]; a1 += xv * rw[1]; a2 += xv * rw[2]; a3 += xv * rw[3];
    a4 += xv * score_w[k * NCLS + c];
  }
  red[tid] = a0; red[64 + tid] = a1; red[128 + tid] = a2;
  red[192 + tid] = a3; red[256 + tid] = a4;
  __syncthreads();
  if (tid < 5) {
    float s = 0;
    for (int j = 0; j < 64; ++j) s += red[tid * 64 + j];
    red[320 + tid] = s;
  }
  __syncthreads();
  if (tid == 0) {
    float r0 = red[320] + reg_b[c0 + 0];
    float r1 = red[321] + reg_b[c0 + 1];
    float r2 = red[322] + reg_b[c0 + 2];
    float r3 = red[323] + reg_b[c0 + 3];
    float sc = red[324] + score_b[c];
    float b0 = nbb[m * 4 + 0], b1 = nbb[m * 4 + 1];
    float b2 = nbb[m * 4 + 2], b3 = nbb[m * 4 + 3];
    float bh = b2 - b0, bw = b3 - b1;
    float bcy = 0.5f * (b0 + b2), bcx = 0.5f * (b1 + b3);
    float ncy = bcy + r0 * bh, ncx = bcx + r1 * bw;
    float nh = bh * expf(r2), nw = bw * expf(r3);
    out[8192 + m * 4 + 0] = r0;
    out[8192 + m * 4 + 1] = r1;
    out[8192 + m * 4 + 2] = r2;
    out[8192 + m * 4 + 3] = r3;
    out[16384 + m] = sc;
    out[18432 + m * 4 + 0] = ncy - nh * 0.5f;
    out[18432 + m * 4 + 1] = ncx - nw * 0.5f;
    out[18432 + m * 4 + 2] = ncy + nh * 0.5f;
    out[18432 + m * 4 + 3] = ncx + nw * 0.5f;
  }
}

// ---------------------------------------------------------------------------
// launch
// ---------------------------------------------------------------------------
extern "C" void kernel_launch(void* const* d_in, const int* in_sizes, int n_in,
                              void* d_out, int out_size, void* d_ws, size_t ws_size,
                              hipStream_t stream) {
  (void)in_sizes; (void)n_in; (void)out_size; (void)ws_size;
  const float* features = (const float*)d_in[0];
  const int*   labels   = (const int*)d_in[1];
  const float* sizes_m  = (const float*)d_in[2];
  const float* weights  = (const float*)d_in[3];
  const int*   cls      = (const int*)d_in[4];
  const float* conv1_w  = (const float*)d_in[5];
  const float* conv1_b  = (const float*)d_in[6];
  const float* conv2_w  = (const float*)d_in[7];
  const float* conv2_b  = (const float*)d_in[8];
  const float* fc1_w    = (const float*)d_in[9];
  const float* fc1_b    = (const float*)d_in[10];
  const float* fc2_w    = (const float*)d_in[11];
  const float* fc2_b    = (const float*)d_in[12];
  const float* reg_w    = (const float*)d_in[13];
  const float* reg_b    = (const float*)d_in[14];
  const float* score_w  = (const float*)d_in[15];
  const float* score_b  = (const float*)d_in[16];
  float* out = (float*)d_out;

  // ---- workspace carve-up (256B aligned) ----
  char* base = (char*)d_ws;
  size_t off = 0;
  auto alloc = [&](size_t bytes) -> char* {
    char* p = base + off;
    off = (off + bytes + 255) & ~(size_t)255;
    return p;
  };
  const int S = Bn * NPn;                       // 2048 segments
  int* mny = (int*)alloc(S * 4);
  int* mnx = (int*)alloc(S * 4);
  int* mxy = (int*)alloc(S * 4);
  int* mxx = (int*)alloc(S * 4);
  unsigned long long* swy = (unsigned long long*)alloc(S * 8);
  unsigned long long* swx = (unsigned long long*)alloc(S * 8);
  unsigned long long* swt = (unsigned long long*)alloc(S * 8);
  float* nbb = (float*)alloc((size_t)S * 4 * 4);
  _Float16* w1h  = (_Float16*)alloc((size_t)K1dim * N1dim * 2);
  _Float16* w2h  = (_Float16*)alloc((size_t)K2dim * N2dim * 2);
  _Float16* fw1h = (_Float16*)alloc((size_t)KF1 * NFC * 2);
  _Float16* fw2h = (_Float16*)alloc((size_t)NFC * NFC * 2);
  _Float16* crops = (_Float16*)alloc((size_t)Mrows * CROPn * CROPn * Cn * 2);
  _Float16* col1  = (_Float16*)alloc((size_t)R1rows * K1dim * 2);
  _Float16* c1out = (_Float16*)alloc((size_t)R1rows * N1dim * 2);
  _Float16* col2  = (_Float16*)alloc((size_t)R2rows * K2dim * 2);
  _Float16* c2out = (_Float16*)alloc((size_t)R2rows * N2dim * 2);  // == fc1 input [2048,2048]
  _Float16* f1out = (_Float16*)alloc((size_t)Mrows * NFC * 2);
  _Float16* f2out = (_Float16*)alloc((size_t)Mrows * NFC * 2);

  // ---- pipeline ----
  ws_init_kernel<<<8, 256, 0, stream>>>(mny, mnx, mxy, mxx, swy, swx, swt);
  seg_reduce_kernel<<<(Bn * Hn * Wn) / 256, 256, 0, stream>>>(
      labels, sizes_m, weights, mny, mnx, mxy, mxx, swy, swx, swt);
  boxes_kernel<<<8, 256, 0, stream>>>(mny, mnx, mxy, mxx, swy, swx, swt, nbb, out);

  cvt_f32_f16_kernel<<<(K1dim * N1dim + 255) / 256, 256, 0, stream>>>(conv1_w, w1h, K1dim * N1dim);
  cvt_f32_f16_kernel<<<(K2dim * N2dim + 255) / 256, 256, 0, stream>>>(conv2_w, w2h, K2dim * N2dim);
  cvt_f32_f16_kernel<<<(KF1 * NFC + 255) / 256, 256, 0, stream>>>(fc1_w, fw1h, KF1 * NFC);
  cvt_f32_f16_kernel<<<(NFC * NFC + 255) / 256, 256, 0, stream>>>(fc2_w, fw2h, NFC * NFC);

  crop_kernel<<<Mrows, 256, 0, stream>>>(features, nbb, crops);

  im2col1_kernel<<<(R1rows * 9 + 255) / 256, 256, 0, stream>>>(crops, col1);
  gemm_f16_wmma<64><<<dim3(R1rows / 128, N1dim / 64), 256, 0, stream>>>(
      col1, w1h, conv1_b, c1out, R1rows, N1dim, K1dim, 1);

  im2col2_kernel<<<(R2rows * 9 + 255) / 256, 256, 0, stream>>>(c1out, col2);
  gemm_f16_wmma<128><<<dim3(R2rows / 128, N2dim / 128), 256, 0, stream>>>(
      col2, w2h, conv2_b, c2out, R2rows, N2dim, K2dim, 1);

  // fc1: c2out memory layout [crop][pos][ch] == [2048, 2048] flatten order
  gemm_f16_wmma<128><<<dim3(Mrows / 128, NFC / 128), 256, 0, stream>>>(
      c2out, fw1h, fc1_b, f1out, Mrows, NFC, KF1, 1);
  gemm_f16_wmma<128><<<dim3(Mrows / 128, NFC / 128), 256, 0, stream>>>(
      f1out, fw2h, fc2_b, f2out, Mrows, NFC, NFC, 1);

  heads_kernel<<<Mrows, 64, 0, stream>>>(
      f2out, reg_w, reg_b, score_w, score_b, cls, nbb, out);
}
